// MultiHeadAttention_1271310320441
// MI455X (gfx1250) — matrix operand order repaired and checked by reading, not compiled, not run on hardware
//
#include <hip/hip_runtime.h>
#include <hip/hip_bf16.h>

// ---------------------------------------------------------------------------
// MHA on gfx1250 (MI455X): all matmuls via v_wmma_f32_16x16x32_f16 (wave32),
// W-tile staging via the Tensor Data Mover (tensor_load_to_lds, TENSORcnt).
// ---------------------------------------------------------------------------

typedef __attribute__((ext_vector_type(16))) _Float16 v16h;
typedef __attribute__((ext_vector_type(8)))  _Float16 v8h;
typedef __attribute__((ext_vector_type(4)))  _Float16 v4h;
typedef __attribute__((ext_vector_type(8)))  float    v8f;
typedef unsigned int u32x4 __attribute__((ext_vector_type(4)));
typedef int          i32x4 __attribute__((ext_vector_type(4)));
typedef int          i32x8 __attribute__((ext_vector_type(8)));

#define D_MODEL 1024
#define SEQ     2048
#define BATCH   2
#define NHEAD   16
#define DKH     64
#define MTOT    (BATCH * SEQ)   // 4096

#ifndef USE_TDM
# if defined(__has_builtin)
#  if __has_builtin(__builtin_amdgcn_tensor_load_to_lds) && \
      __has_builtin(__builtin_amdgcn_s_wait_tensorcnt)
#   define USE_TDM 1
#  else
#   define USE_TDM 0
#  endif
# else
#  define USE_TDM 0
# endif
#endif

__device__ __forceinline__ v16h combine16(v8h lo, v8h hi) {
    return __builtin_shufflevector(lo, hi, 0,1,2,3,4,5,6,7,8,9,10,11,12,13,14,15);
}

__device__ __forceinline__ v8f wmma_f16(v16h a, v16h b, v8f c) {
    // (neg_a, A, neg_b, B, c_mod, C, reuse_a, reuse_b)
    return __builtin_amdgcn_wmma_f32_16x16x32_f16(false, a, false, b, (short)0, c,
                                                  false, false);
}

#if USE_TDM
// TDM descriptor: 2D tile of 64 rows x 32 f16, row stride = D_MODEL elements,
// destination LDS packed [row][32]. Bitfields per CDNA5 ISA.
// Toolchain uses the 6-arg builtin: (g0, g1, g2, g3, g4, cpol).
__device__ __forceinline__ void tdm_load_tile_64x32(const _Float16* gsrc,
                                                    _Float16* lds_dst) {
    unsigned long long ga = (unsigned long long)(size_t)gsrc;     // byte address
    unsigned int       la = (unsigned int)(size_t)lds_dst;        // LDS offset = addr[31:0]
    u32x4 g0 = { 1u,                                   // count=1 (valid descriptor)
                 la,                                   // lds_addr
                 (unsigned int)ga,                     // global_addr[31:0]
                 (unsigned int)((ga >> 32) & 0x01FFFFFFu) | 0x80000000u }; // [56:32]|type=2
    i32x8 g1 = { (1 << 16),              // data_size = 2 bytes
                 (D_MODEL << 16),        // tensor_dim0 (low 16 @ bits 63:48)
                 (D_MODEL << 16),        // tensor_dim1 (low 16 @ bits 95:80)
                 (32 << 16),             // tile_dim0 = 32  (bits 127:112)
                 64,                     // tile_dim1 = 64  (bits 143:128)
                 D_MODEL,                // tensor_dim0_stride low32 (bits 191:160)
                 0, 0 };
    i32x4 z4 = {0, 0, 0, 0};
    i32x8 z8 = {0, 0, 0, 0, 0, 0, 0, 0};
    __builtin_amdgcn_tensor_load_to_lds(g0, g1, z4, z4, z8, 0);
}
#endif

// ---------------------------------------------------------------------------
// fp32 -> fp16 conversion (4 elements / thread, sizes are multiples of 1024)
// ---------------------------------------------------------------------------
__global__ void cvt_f32_to_f16(const float* __restrict__ in,
                               _Float16* __restrict__ out, int n) {
    int i = (blockIdx.x * blockDim.x + threadIdx.x) * 4;
    if (i < n) {
        float4 v = *(const float4*)(in + i);
        v4h o = { (_Float16)v.x, (_Float16)v.y, (_Float16)v.z, (_Float16)v.w };
        *(v4h*)(out + i) = o;
    }
}

// ---------------------------------------------------------------------------
// GEMM: Y[m,n] = sum_k A[m,k] * W[n,k] + bias[n]   (i.e. X @ W^T + b)
// Block: 256 threads = 8 waves. Block tile: 128 (M) x 64 (N). K step = 32.
// W tile double-buffered in LDS via TDM (wave 0 is the DMA leader).
// mode 0: store f16 head-split [B, H, S, 64]; mode 1: store f32 flat.
// ---------------------------------------------------------------------------
__global__ void gemm_xwt(const _Float16* __restrict__ A,
                         const _Float16* __restrict__ W,
                         const float* __restrict__ bias,
                         _Float16* __restrict__ outH,
                         float* __restrict__ outF,
                         int mode) {
    __shared__ __align__(64) _Float16 BtBuf[2 * 64 * 32];  // double buffer

    const int wave = threadIdx.x >> 5;   // 0..7
    const int lane = threadIdx.x & 31;
    const int ln16 = lane & 15;
    const int hi   = lane >> 4;          // 0: lanes 0-15, 1: lanes 16-31

    const int m0 = blockIdx.x * 128 + wave * 16;  // wave's 16 output rows
    const int n0 = blockIdx.y * 64;               // block's 64 output cols

    v8f acc[4] = { {0,0,0,0,0,0,0,0}, {0,0,0,0,0,0,0,0},
                   {0,0,0,0,0,0,0,0}, {0,0,0,0,0,0,0,0} };

    const _Float16* Arow  = A + (size_t)(m0 + ln16) * D_MODEL;
    const _Float16* Wtile = W + (size_t)n0 * D_MODEL;

#if USE_TDM
    // Scalar (SGPR) branch: TDM ignores EXEC, so only wave 0's instruction
    // stream may contain the descriptor issue.
    const bool leader = (__builtin_amdgcn_readfirstlane((int)threadIdx.x) < 32);
    if (leader) tdm_load_tile_64x32(Wtile, BtBuf);

    for (int i = 0; i < D_MODEL / 32; ++i) {
        const int k0 = i * 32;
        if (leader) __builtin_amdgcn_s_wait_tensorcnt(0);
        __syncthreads();   // buffer (i&1) filled AND previous reads of (1-i&1) done
        if (leader && (i + 1) < D_MODEL / 32)
            tdm_load_tile_64x32(Wtile + (size_t)(k0 + 32),
                                BtBuf + ((i + 1) & 1) * (64 * 32));
        const _Float16* Bt = BtBuf + (i & 1) * (64 * 32);

        if (k0 + 32 < D_MODEL) __builtin_prefetch(Arow + k0 + 32, 0, 0);

        // A fragment (16x32, contiguous k = 16*hi + s, same k order as B)
        v8h a_lo = *(const v8h*)(Arow + k0 + hi * 16);
        v8h a_hi = *(const v8h*)(Arow + k0 + hi * 16 + 8);
        v16h a = combine16(a_lo, a_hi);

        // Preload all 4 B fragments, then burst the 4 WMMAs.
        v16h b[4];
        #pragma unroll
        for (int nt = 0; nt < 4; ++nt)
            b[nt] = *(const v16h*)(Bt + (nt * 16 + ln16) * 32 + hi * 16);
        #pragma unroll
        for (int nt = 0; nt < 4; ++nt)
            acc[nt] = wmma_f16(a, b[nt], acc[nt]);
    }
#else
    _Float16* Bt = BtBuf;
    for (int k0 = 0; k0 < D_MODEL; k0 += 32) {
        {
            const int t  = threadIdx.x;
            const int nl = t >> 2;
            const int kl = (t & 3) * 8;
            *(v8h*)(Bt + nl * 32 + kl) =
                *(const v8h*)(Wtile + (size_t)nl * D_MODEL + k0 + kl);
        }
        __syncthreads();
        if (k0 + 32 < D_MODEL) __builtin_prefetch(Arow + k0 + 32, 0, 0);
        v8h a_lo = *(const v8h*)(Arow + k0 + hi * 16);
        v8h a_hi = *(const v8h*)(Arow + k0 + hi * 16 + 8);
        v16h a = combine16(a_lo, a_hi);
        v16h b[4];
        #pragma unroll
        for (int nt = 0; nt < 4; ++nt)
            b[nt] = *(const v16h*)(Bt + (nt * 16 + ln16) * 32 + hi * 16);
        #pragma unroll
        for (int nt = 0; nt < 4; ++nt)
            acc[nt] = wmma_f16(a, b[nt], acc[nt]);
        __syncthreads();
    }
#endif

    // ---- epilogue: C-layout = lane holds column n, rows hi*8+r ----
    #pragma unroll
    for (int nt = 0; nt < 4; ++nt) {
        const int n  = n0 + nt * 16 + ln16;
        const float bv = bias[n];
        #pragma unroll
        for (int r = 0; r < 8; ++r) {
            const int m = m0 + hi * 8 + r;
            const float v = acc[nt][r] + bv;
            if (mode == 0) {
                const int b = m >> 11;
                const int s = m & 2047;
                const int h = n >> 6;
                const int d = n & 63;
                outH[(((size_t)b * NHEAD + h) * SEQ + s) * DKH + d] = (_Float16)v;
            } else {
                outF[(size_t)m * D_MODEL + n] = v;
            }
        }
    }
}

// ---------------------------------------------------------------------------
// Flash attention: one block = 4 waves sharing (b,h); each wave owns 16 queries.
// qp/kp/vp: [B*H, S, 64] f16.  attn out: [B*S, 1024] f16 (head-merged).
// ---------------------------------------------------------------------------
__global__ void attention_fwd(const _Float16* __restrict__ qp,
                              const _Float16* __restrict__ kp,
                              const _Float16* __restrict__ vp,
                              _Float16* __restrict__ attn) {
    __shared__ __align__(64) _Float16 Vt[64 * 32];      // [d][key] transposed V tile
    __shared__ __align__(64) _Float16 Pb[4][16 * 32];   // per-wave P (row-major)

    const int wave = threadIdx.x >> 5;   // 0..3
    const int lane = threadIdx.x & 31;
    const int ln16 = lane & 15;
    const int hi   = lane >> 4;

    const int bh    = blockIdx.x;                    // 0..31  (b*16 + h)
    const int qbase = blockIdx.y * 64 + wave * 16;

    const _Float16* Kbase = kp + (size_t)bh * SEQ * DKH;
    const _Float16* Vbase = vp + (size_t)bh * SEQ * DKH;

    // ---- Q A-fragments (16 x 64 = two 16x32 fragments), kept in regs ----
    const _Float16* qrow = qp + ((size_t)bh * SEQ + qbase + ln16) * DKH;
    v16h qa[2];
    #pragma unroll
    for (int kc = 0; kc < 2; ++kc) {
        v8h lo = *(const v8h*)(qrow + kc * 32 + hi * 16);
        v8h h8 = *(const v8h*)(qrow + kc * 32 + hi * 16 + 8);
        qa[kc] = combine16(lo, h8);
    }

    v8f o[4] = { {0,0,0,0,0,0,0,0}, {0,0,0,0,0,0,0,0},
                 {0,0,0,0,0,0,0,0}, {0,0,0,0,0,0,0,0} };
    float mrow[8], lrow[8];
    #pragma unroll
    for (int r = 0; r < 8; ++r) { mrow[r] = -1e30f; lrow[r] = 0.0f; }

    const float scale = 0.125f;  // 1/sqrt(64)

    for (int j = 0; j < SEQ; j += 32) {
        // ---- cooperative transpose-stage of V tile (32 keys x 64 dims) ----
        {
            const int t   = threadIdx.x;
            const int key = t >> 2;              // 0..31
            const int d0  = (t & 3) * 16;        // 0,16,32,48
            const _Float16* src = Vbase + (size_t)(j + key) * DKH + d0;
            v8h v0 = *(const v8h*)(src);
            v8h v1 = *(const v8h*)(src + 8);
            #pragma unroll
            for (int e = 0; e < 8; ++e) Vt[(d0 + e)     * 32 + key] = v0[e];
            #pragma unroll
            for (int e = 0; e < 8; ++e) Vt[(d0 + 8 + e) * 32 + key] = v1[e];
        }
        __syncthreads();

        // ---- scores: preload K^T fragments, then WMMA burst ----
        v16h kb[2][2];
        #pragma unroll
        for (int g = 0; g < 2; ++g)
            #pragma unroll
            for (int kc = 0; kc < 2; ++kc)
                kb[g][kc] = *(const v16h*)(Kbase +
                            (size_t)(j + g * 16 + ln16) * DKH + kc * 32 + hi * 16);
        v8f sc[2];
        #pragma unroll
        for (int g = 0; g < 2; ++g) {
            v8f s = {0,0,0,0,0,0,0,0};
            s = wmma_f16(qa[0], kb[g][0], s);
            s = wmma_f16(qa[1], kb[g][1], s);
            sc[g] = s * scale;
        }

        // ---- online softmax (row-wise over 32 keys; rows per lane-half) ----
        float mnew[8], alpha[8];
        #pragma unroll
        for (int r = 0; r < 8; ++r) {
            float mx = fmaxf(sc[0][r], sc[1][r]);
            #pragma unroll
            for (int off = 1; off < 16; off <<= 1)
                mx = fmaxf(mx, __shfl_xor(mx, off, 32));
            mnew[r]  = fmaxf(mrow[r], mx);
            alpha[r] = __expf(mrow[r] - mnew[r]);
            mrow[r]  = mnew[r];
        }

        // ---- P = exp(S - m); row-sum; spill P to LDS row-major ----
        #pragma unroll
        for (int r = 0; r < 8; ++r) {
            float p0 = __expf(sc[0][r] - mnew[r]);
            float p1 = __expf(sc[1][r] - mnew[r]);
            float ps = p0 + p1;
            #pragma unroll
            for (int off = 1; off < 16; off <<= 1)
                ps += __shfl_xor(ps, off, 32);
            lrow[r] = lrow[r] * alpha[r] + ps;
            const int row = hi * 8 + r;                 // C-layout rows
            Pb[wave][row * 32 + ln16]      = (_Float16)p0;
            Pb[wave][row * 32 + 16 + ln16] = (_Float16)p1;
        }
        asm volatile("s_wait_dscnt 0" ::: "memory");

        // ---- rescale running output ----
        #pragma unroll
        for (int dt = 0; dt < 4; ++dt)
            #pragma unroll
            for (int r = 0; r < 8; ++r)
                o[dt][r] *= alpha[r];

        // ---- reload P as A-fragment; preload V fragments; WMMA burst ----
        const _Float16* prow = &Pb[wave][ln16 * 32];
        v16h pa = combine16(*(const v8h*)(prow + hi * 16),
                            *(const v8h*)(prow + hi * 16 + 8));
        v16h vbf[4];
        #pragma unroll
        for (int dt = 0; dt < 4; ++dt)
            vbf[dt] = *(const v16h*)(Vt + (dt * 16 + ln16) * 32 + hi * 16);
        #pragma unroll
        for (int dt = 0; dt < 4; ++dt)
            o[dt] = wmma_f16(pa, vbf[dt], o[dt]);
        __syncthreads();
    }

    // ---- finalize: O / l, store head-merged f16 [B*S, 1024] ----
    const int b = bh >> 4;
    const int h = bh & 15;
    #pragma unroll
    for (int dt = 0; dt < 4; ++dt) {
        const int d = h * DKH + dt * 16 + ln16;
        #pragma unroll
        for (int r = 0; r < 8; ++r) {
            const int s = qbase + hi * 8 + r;
            const float v = o[dt][r] / lrow[r];
            attn[((size_t)b * SEQ + s) * D_MODEL + d] = (_Float16)v;
        }
    }
}

// ---------------------------------------------------------------------------
// Host launcher
// ---------------------------------------------------------------------------
extern "C" void kernel_launch(void* const* d_in, const int* in_sizes, int n_in,
                              void* d_out, int out_size, void* d_ws, size_t ws_size,
                              hipStream_t stream) {
    const float* Q  = (const float*)d_in[0];
    const float* K  = (const float*)d_in[1];
    const float* V  = (const float*)d_in[2];
    const float* Wq = (const float*)d_in[3];
    const float* bq = (const float*)d_in[4];
    const float* Wk = (const float*)d_in[5];
    const float* bk = (const float*)d_in[6];
    const float* Wv = (const float*)d_in[7];
    const float* bv = (const float*)d_in[8];
    const float* Wo = (const float*)d_in[9];
    const float* bo = (const float*)d_in[10];
    float* out = (float*)d_out;

    const size_t NX = (size_t)MTOT * D_MODEL;       // 4,194,304
    const size_t NW = (size_t)D_MODEL * D_MODEL;    // 1,048,576

    _Float16* w = (_Float16*)d_ws;
    _Float16* Qh  = w; w += NX;
    _Float16* Kh  = w; w += NX;
    _Float16* Vh  = w; w += NX;
    _Float16* Wqh = w; w += NW;
    _Float16* Wkh = w; w += NW;
    _Float16* Wvh = w; w += NW;
    _Float16* Woh = w; w += NW;
    _Float16* qpj = w; w += NX;   // [B,H,S,64]
    _Float16* kpj = w; w += NX;
    _Float16* vpj = w; w += NX;
    _Float16* ath = w; w += NX;   // [B*S, 1024]

    cvt_f32_to_f16<<<(int)(NX / 1024), 256, 0, stream>>>(Q, Qh, (int)NX);
    cvt_f32_to_f16<<<(int)(NX / 1024), 256, 0, stream>>>(K, Kh, (int)NX);
    cvt_f32_to_f16<<<(int)(NX / 1024), 256, 0, stream>>>(V, Vh, (int)NX);
    cvt_f32_to_f16<<<(int)(NW / 1024), 256, 0, stream>>>(Wq, Wqh, (int)NW);
    cvt_f32_to_f16<<<(int)(NW / 1024), 256, 0, stream>>>(Wk, Wkh, (int)NW);
    cvt_f32_to_f16<<<(int)(NW / 1024), 256, 0, stream>>>(Wv, Wvh, (int)NW);
    cvt_f32_to_f16<<<(int)(NW / 1024), 256, 0, stream>>>(Wo, Woh, (int)NW);

    dim3 ggrid(MTOT / 128, D_MODEL / 64);   // (32, 16)
    gemm_xwt<<<ggrid, 256, 0, stream>>>(Qh, Wqh, bq, qpj, nullptr, 0);
    gemm_xwt<<<ggrid, 256, 0, stream>>>(Kh, Wkh, bk, kpj, nullptr, 0);
    gemm_xwt<<<ggrid, 256, 0, stream>>>(Vh, Wvh, bv, vpj, nullptr, 0);

    attention_fwd<<<dim3(BATCH * NHEAD, SEQ / 64), 128, 0, stream>>>(qpj, kpj, vpj, ath);

    gemm_xwt<<<ggrid, 256, 0, stream>>>(ath, Woh, bo, nullptr, out, 1);
}
